// FlattenedVectorQuantizer_10239202034429
// MI455X (gfx1250) — matrix operand order repaired and checked by reading, not compiled, not run on hardware
//
#include <hip/hip_runtime.h>
#include <hip/hip_bf16.h>
#include <float.h>

typedef __attribute__((ext_vector_type(16))) __bf16          v16bf;
typedef __attribute__((ext_vector_type(8)))  float           v8f;
typedef __attribute__((ext_vector_type(16))) unsigned short  v16us;
typedef __attribute__((ext_vector_type(8)))  unsigned int    v8u;
typedef __attribute__((ext_vector_type(4)))  int             v4i;

typedef __attribute__((address_space(1))) v4i* gv4p;   // global int4*
typedef __attribute__((address_space(3))) v4i* lv4p;   // LDS    int4*

#define K_CODES 4096
#define DDIM    256
#define NROWS   65536
#define NGROUPS (K_CODES / 16)     // 256 groups of 16 codes
#define TILE_SH (16 * DDIM)        // one B tile: 16 codes x 256 d (bf16) = 8KB

#if defined(__gfx1250__) && __has_builtin(__builtin_amdgcn_global_load_async_to_lds_b128)
#define VQ_ASYNC 1
#else
#define VQ_ASYNC 0
#endif

__device__ __forceinline__ unsigned short f2bf(float f) {
    // round-to-nearest-even f32 -> bf16
    unsigned int u = __builtin_bit_cast(unsigned int, f);
    u += 0x7FFFu + ((u >> 16) & 1u);
    return (unsigned short)(u >> 16);
}

#if VQ_ASYNC
__device__ __forceinline__ void vq_async_b128(const void* g, void* l) {
    void* gnc = const_cast<void*>(g);
    __builtin_amdgcn_global_load_async_to_lds_b128((gv4p)gnc, (lv4p)l, 0, 0);
}
__device__ __forceinline__ void vq_wait_async() {
#if __has_builtin(__builtin_amdgcn_s_wait_asynccnt)
    __builtin_amdgcn_s_wait_asynccnt(0);
#else
    asm volatile("s_wait_asynccnt 0" ::: "memory");
#endif
}
#endif

// ---------------------------------------------------------------------------
// Prep: emb f32 -> bf16 copy (L2-resident), and |e_k|^2 per code (f32).
// ---------------------------------------------------------------------------
__global__ void __launch_bounds__(256) vq_prep(const float* __restrict__ emb,
                                               unsigned short* __restrict__ embb,
                                               float* __restrict__ enorm) {
    __shared__ float red[256];
    const int k = blockIdx.x;
    const int t = threadIdx.x;
    float f = emb[k * DDIM + t];
    embb[k * DDIM + t] = f2bf(f);
    red[t] = f * f;
    __syncthreads();
    for (int s = 128; s > 0; s >>= 1) {
        if (t < s) red[t] += red[t + s];
        __syncthreads();
    }
    if (t == 0) enorm[k] = red[0];
}

// ---------------------------------------------------------------------------
// Main argmin kernel.
//  - block = 8 waves; wave owns 32 z-rows (two 16-row A fragment sets, held
//    in VGPRs for the whole codebook sweep).
//  - the 16-code x 256-d bf16 B tile (8KB) is staged in LDS once per block
//    and shared by all 8 waves; double-buffered, prefetched with
//    global_load_async_to_lds_b128 when available.
//  - per chunk: one LDS B fragment feeds two independent WMMAs.
// ---------------------------------------------------------------------------
__global__ void __launch_bounds__(256) vq_argmin(const float* __restrict__ z,
                                                 const unsigned short* __restrict__ embb,
                                                 const float* __restrict__ enorm,
                                                 int* __restrict__ outIdx) {
    __shared__ alignas(16) unsigned short tiles[2][TILE_SH];   // 2 x 8KB

    const int t    = threadIdx.x;
    const int wave = t >> 5;
    const int lane = t & 31;
    const int half = lane >> 4;
    const int l15  = lane & 15;
    const int rowBase = (blockIdx.x * 8 + wave) * 32;

    // ---- A fragments (16-bit A 16x32 layout): two 16-row tiles
    v16bf a0[8], a1[8];
    #pragma unroll
    for (int rt = 0; rt < 2; ++rt) {
        const float* zr = z + (size_t)(rowBase + rt * 16 + l15) * DDIM;
        #pragma unroll
        for (int c = 0; c < 8; ++c) {
            const int kb1 = 32 * c + (half ? 8 : 0);
            const float4* p1 = (const float4*)(zr + kb1);
            const float4* p2 = (const float4*)(zr + kb1 + 16);
            float4 x0 = p1[0], x1 = p1[1];
            float4 y0 = p2[0], y1 = p2[1];
            v16us au;
            au[0]  = f2bf(x0.x); au[1]  = f2bf(x0.y); au[2]  = f2bf(x0.z); au[3]  = f2bf(x0.w);
            au[4]  = f2bf(x1.x); au[5]  = f2bf(x1.y); au[6]  = f2bf(x1.z); au[7]  = f2bf(x1.w);
            au[8]  = f2bf(y0.x); au[9]  = f2bf(y0.y); au[10] = f2bf(y0.z); au[11] = f2bf(y0.w);
            au[12] = f2bf(y1.x); au[13] = f2bf(y1.y); au[14] = f2bf(y1.z); au[15] = f2bf(y1.w);
            if (rt == 0) a0[c] = __builtin_bit_cast(v16bf, au);
            else         a1[c] = __builtin_bit_cast(v16bf, au);
        }
    }

    float best0[8], best1[8];
    int   bidx0[8], bidx1[8];
    #pragma unroll
    for (int j = 0; j < 8; ++j) {
        best0[j] = FLT_MAX; bidx0[j] = 0;
        best1[j] = FLT_MAX; bidx1[j] = 0;
    }

    // ---- stage tile 0 (each thread copies 32B of the contiguous 8KB tile)
#if VQ_ASYNC
    {
        const char* gsrc = (const char*)embb + (size_t)t * 32;
        char*       ldst = (char*)&tiles[0][0] + (size_t)t * 32;
        vq_async_b128(gsrc,      ldst);
        vq_async_b128(gsrc + 16, ldst + 16);
        vq_wait_async();
    }
#else
    {
        const uint4* src = (const uint4*)embb;
        uint4 v0 = src[t * 2], v1 = src[t * 2 + 1];
        uint4* dst = (uint4*)&tiles[0][0];
        dst[t * 2] = v0; dst[t * 2 + 1] = v1;
    }
#endif
    __syncthreads();

    // ---- sweep the codebook
    for (int g = 0; g < NGROUPS; ++g) {
        const int buf = g & 1;
        const int k0  = g * 16;
        const int code = k0 + l15;

        // prefetch next tile into the other buffer
#if VQ_ASYNC
        if (g + 1 < NGROUPS) {
            const char* gsrc = (const char*)(embb + (size_t)(g + 1) * TILE_SH) + (size_t)t * 32;
            char*       ldst = (char*)&tiles[buf ^ 1][0] + (size_t)t * 32;
            vq_async_b128(gsrc,      ldst);
            vq_async_b128(gsrc + 16, ldst + 16);
        }
#else
        uint4 pr0, pr1;
        if (g + 1 < NGROUPS) {
            const uint4* src = (const uint4*)(embb + (size_t)(g + 1) * TILE_SH);
            pr0 = src[t * 2]; pr1 = src[t * 2 + 1];
        }
#endif

        const float en = enorm[code];
        v8f acc0, acc1;
        #pragma unroll
        for (int j = 0; j < 8; ++j) { acc0[j] = 0.0f; acc1[j] = 0.0f; }

        // B 32x16 layout: lane = column; lanes 0-15 K=0..7,16..23, lanes 16-31
        // K=8..15,24..31 of each 32-chunk -> two 16B ds loads per chunk.
        const unsigned short* bt = &tiles[buf][l15 * DDIM + (half ? 8 : 0)];
        #pragma unroll
        for (int c = 0; c < 8; ++c) {
            const uint4* p = (const uint4*)(bt + 32 * c);
            uint4 lo = p[0], hi = p[2];
            v8u bu;
            bu[0] = lo.x; bu[1] = lo.y; bu[2] = lo.z; bu[3] = lo.w;
            bu[4] = hi.x; bu[5] = hi.y; bu[6] = hi.z; bu[7] = hi.w;
            v16bf b = __builtin_bit_cast(v16bf, bu);
            acc0 = __builtin_amdgcn_wmma_f32_16x16x32_bf16(
                false, a0[c], false, b, (short)0, acc0, false, false);
            acc1 = __builtin_amdgcn_wmma_f32_16x16x32_bf16(
                false, a1[c], false, b, (short)0, acc1, false, false);
        }

        #pragma unroll
        for (int j = 0; j < 8; ++j) {
            float s0 = en - 2.0f * acc0[j];     // |z|^2 dropped: row-constant
            if (s0 < best0[j]) { best0[j] = s0; bidx0[j] = code; }
            float s1 = en - 2.0f * acc1[j];
            if (s1 < best1[j]) { best1[j] = s1; bidx1[j] = code; }
        }

        // commit next tile / make it visible before the next iteration
#if VQ_ASYNC
        vq_wait_async();
#else
        if (g + 1 < NGROUPS) {
            uint4* dst = (uint4*)&tiles[buf ^ 1][0];
            dst[t * 2] = pr0; dst[t * 2 + 1] = pr1;
        }
#endif
        __syncthreads();
    }

    // ---- reduce across the 16 lanes sharing each row (xor stays in-half)
    #pragma unroll
    for (int j = 0; j < 8; ++j) {
        float s0 = best0[j]; int i0 = bidx0[j];
        float s1 = best1[j]; int i1 = bidx1[j];
        #pragma unroll
        for (int off = 1; off < 16; off <<= 1) {
            float os0 = __shfl_xor(s0, off, 32);
            int   oi0 = __shfl_xor(i0, off, 32);
            if (os0 < s0 || (os0 == s0 && oi0 < i0)) { s0 = os0; i0 = oi0; }
            float os1 = __shfl_xor(s1, off, 32);
            int   oi1 = __shfl_xor(i1, off, 32);
            if (os1 < s1 || (os1 == s1 && oi1 < i1)) { s1 = os1; i1 = oi1; }
        }
        bidx0[j] = i0; bidx1[j] = i1;
    }

    // C/D layout: VGPR j holds row M=j (lanes 0-15) or M=j+8 (lanes 16-31)
    if (l15 == 0) {
        #pragma unroll
        for (int j = 0; j < 8; ++j) {
            outIdx[rowBase + half * 8 + j]      = bidx0[j];
            outIdx[rowBase + 16 + half * 8 + j] = bidx1[j];
        }
    }
}

// ---------------------------------------------------------------------------
// Gather: quantized_st = emb[idx] (straight-through value == gather),
// per-row squared-error partial for the loss, indices emitted as float.
// ---------------------------------------------------------------------------
__global__ void __launch_bounds__(256) vq_gather(const float* __restrict__ z,
                                                 const float* __restrict__ emb,
                                                 const int* __restrict__ idx,
                                                 float* __restrict__ outQ,
                                                 float* __restrict__ outI,
                                                 float* __restrict__ partial) {
    __shared__ float red[256];
    const int row = blockIdx.x;
    const int t   = threadIdx.x;
    const int k   = idx[row];
    float q  = emb[(size_t)k * DDIM + t];
    float zz = z[(size_t)row * DDIM + t];
    outQ[(size_t)row * DDIM + t] = q;
    float d = q - zz;
    red[t] = d * d;
    __syncthreads();
    for (int s = 128; s > 0; s >>= 1) {
        if (t < s) red[t] += red[t + s];
        __syncthreads();
    }
    if (t == 0) { partial[row] = red[0]; outI[row] = (float)k; }
}

// ---------------------------------------------------------------------------
// Finalize: deterministic fixed-order reduction of 65536 row partials.
// loss = q_latent + 0.25*e_latent = 1.25 * MSE (stop_gradient is identity fwd)
// ---------------------------------------------------------------------------
__global__ void __launch_bounds__(256) vq_finalize(const float* __restrict__ partial,
                                                   float* __restrict__ outLoss) {
    __shared__ float red[256];
    const int t = threadIdx.x;
    float s = 0.0f;
    for (int i = t; i < NROWS; i += 256) s += partial[i];
    red[t] = s;
    __syncthreads();
    for (int k = 128; k > 0; k >>= 1) {
        if (t < k) red[t] += red[t + k];
        __syncthreads();
    }
    if (t == 0) outLoss[0] = 1.25f * red[0] / (float)((long)NROWS * DDIM);
}

extern "C" void kernel_launch(void* const* d_in, const int* in_sizes, int n_in,
                              void* d_out, int out_size, void* d_ws, size_t ws_size,
                              hipStream_t stream) {
    const float* z   = (const float*)d_in[0];   // [65536, 256] f32
    const float* emb = (const float*)d_in[1];   // [4096, 256]  f32

    float* out     = (float*)d_out;
    float* outLoss = out;                              // [0]
    float* outQ    = out + 1;                          // [1 .. 1+N*D)
    float* outI    = out + 1 + (size_t)NROWS * DDIM;   // [.. +N)

    // workspace layout (all 16B-aligned):
    //   [0, 2MB)            emb as bf16
    //   [2MB, 2MB+16KB)     |e_k|^2 f32
    //   [+16KB, +272KB)     argmin indices i32
    //   [+272KB, +528KB)    per-row loss partials f32
    char* ws = (char*)d_ws;
    unsigned short* embb  = (unsigned short*)ws;
    float*          enorm = (float*)(ws + (size_t)2 * 1024 * 1024);
    int*            idx   = (int*)  (ws + (size_t)2 * 1024 * 1024 + 16 * 1024);
    float*          part  = (float*)(ws + (size_t)2 * 1024 * 1024 + 16 * 1024 + 256 * 1024);

    vq_prep    <<<K_CODES,     256, 0, stream>>>(emb, embb, enorm);
    vq_argmin  <<<NROWS / 256, 256, 0, stream>>>(z, embb, enorm, idx);
    vq_gather  <<<NROWS,       256, 0, stream>>>(z, emb, idx, outQ, outI, part);
    vq_finalize<<<1,           256, 0, stream>>>(part, outLoss);
}
